// WindowCrossAttention_61357902791242
// MI455X (gfx1250) — compile-verified, matrix operand order checked by reading, not run on hardware
//
#include <hip/hip_runtime.h>

#define DIMC   384
#define NHEAD  12
#define HDIM   32
#define CDIM   192        // DIM / CS
#define NKV    49
#define NKVP   64
#define QNQ    196
#define NTILE  14         // 16-row query tiles (224 rows padded)
#define GROUPS 7          // 32-row query groups
#define KVOUT  768
#define BWIN   2048
#define SCALE_F 0.17677669529663687f

typedef __attribute__((ext_vector_type(16))) __bf16 v16bf;
typedef __attribute__((ext_vector_type(8)))  float  v8f;

__device__ __forceinline__ v8f wmma_bf16(v16bf a, v16bf b, v8f c) {
  // D = A(16x32) * B(32x16) + C, f32 accumulate
  return __builtin_amdgcn_wmma_f32_16x16x32_bf16(false, a, false, b, (short)0, c, false, false);
}

// ---- prep: bf16-transpose weights + bias in WMMA-accumulator fragment layout ----
__global__ void wca_prep_kernel(const float* __restrict__ w_kv,
                                const float* __restrict__ w_q,
                                const float* __restrict__ w_proj,
                                const float* __restrict__ bias_table,
                                __bf16* __restrict__ w_kvT,     // [768][384]
                                __bf16* __restrict__ w_qT,      // [384][192]
                                __bf16* __restrict__ w_projT,   // [192][384]
                                float*  __restrict__ biasPk) {  // [12][14][32][32]
  int i = blockIdx.x * blockDim.x + threadIdx.x;
  const int n0 = KVOUT * DIMC;              // 294912
  const int n1 = DIMC * CDIM;               // 73728
  const int n2 = CDIM * DIMC;               // 73728
  const int n3 = NHEAD * NTILE * 32 * 32;   // 172032
  if (i < n0) {
    int n = i / DIMC, k = i % DIMC;
    w_kvT[i] = (__bf16)w_kv[k * KVOUT + n];
    return;
  }
  i -= n0;
  if (i < n1) {
    int n = i / CDIM, k = i % CDIM;
    w_qT[i] = (__bf16)w_q[k * DIMC + n];
    return;
  }
  i -= n1;
  if (i < n2) {
    int n = i / DIMC, k = i % DIMC;
    w_projT[i] = (__bf16)w_proj[k * CDIM + n];
    return;
  }
  i -= n2;
  if (i < n3) {
    int h    = i / (NTILE * 1024);
    int rem  = i % (NTILE * 1024);
    int t    = rem / 1024;           // 16-row tile
    int rem2 = rem % 1024;
    int lane = rem2 / 32;
    int j    = rem2 % 32;            // j = nt*8 + r
    int nt = j / 8, r = j % 8;
    int qi = t * 16 + r + 8 * (lane >> 4);
    int ki = nt * 16 + (lane & 15);
    float val = -1.0e30f;            // softmax mask for pad
    if (qi < QNQ && ki < NKV) {
      int qh = (qi / 14) >> 1, qw = (qi % 14) >> 1;
      int jj = 4 * ki;
      int kh = (jj / 14) >> 1, kw = (jj % 14) >> 1;
      int idx = (qh - kh + 6) * 13 + (qw - kw + 6);
      val = bias_table[idx * NHEAD + h];
    }
    biasPk[i] = val;
  }
}

// ---------------- fused attention: one block per window ----------------
__global__ __launch_bounds__(256)
void wca_main_kernel(const float* __restrict__ kv,      // [B][49][384]
                     const float* __restrict__ q,       // [B][196][192]
                     const float* __restrict__ b_kv,    // [768]
                     const float* __restrict__ b_q,     // [384]
                     const float* __restrict__ b_proj,  // [192]
                     const __bf16* __restrict__ w_kvT,  // [768][384]
                     const __bf16* __restrict__ w_qT,   // [384][192]
                     const __bf16* __restrict__ w_projT,// [192][384]
                     const float* __restrict__ biasPk,  // [12][14][32][32]
                     float* __restrict__ out) {         // [B][196][192]
  __shared__ __align__(32) __bf16 smem[81920];          // 160 KB
  __bf16* kmat = smem;                        // [12][64][32]  K per head (row=krow)
  __bf16* vmat = smem + 24576;                // [12][32][64]  V^T per head (row=hd)
  __bf16* kvb  = smem + 49152;                // [64][384]     phase 1-2 only
  __bf16* qpt  = smem + 49152;                // [32][384]     phase 3+
  __bf16* obuf = smem + 49152 + 12288;        // [32][384]
  __bf16* pbuf = smem + 49152 + 24576;        // [8][16][64]   per-wave P
  __bf16* qg   = smem + 49152 + 24576;        // [32][192]     (time-shared w/ pbuf)

  const int b    = blockIdx.x;
  const int tid  = threadIdx.x;
  const int wave = tid >> 5;
  const int lane = tid & 31;
  const int l16  = lane & 15;
  const int hi   = lane >> 4;

  // ---- Phase 1: stage kv[b] -> bf16 LDS, rows 49..63 zeroed ----
  const float* kvsrc = kv + (size_t)b * NKV * DIMC;
  for (int i = tid; i < NKVP * DIMC; i += 256) {
    int r = i / DIMC, c = i % DIMC;
    kvb[i] = (__bf16)((r < NKV) ? kvsrc[r * DIMC + c] : 0.0f);
  }
  __syncthreads();

  // ---- Phase 2: kvp = kvb @ w_kv + b_kv  ->  kmat / vmat(T);  192 tasks, 24/wave ----
  for (int t = wave; t < 4 * 48; t += 8) {
    int m = t / 48, n = t % 48;
    const __bf16* arow = kvb + (m * 16 + l16) * DIMC;
    const __bf16* brow = w_kvT + (n * 16 + l16) * DIMC;
    v8f acc = {};
#pragma unroll
    for (int kc = 0; kc < 12; ++kc) {
      v16bf a  = *(const v16bf*)(arow + kc * 32 + hi * 16);
      v16bf bb = *(const v16bf*)(brow + kc * 32 + hi * 16);
      acc = wmma_bf16(a, bb, acc);
    }
    int ncol = n * 16;
    float bias = b_kv[ncol + l16];
    int pair = ncol / DIMC;           // 0 -> K, 1 -> V
    int h    = (ncol % DIMC) / HDIM;
    int hdb  = ncol % HDIM;           // 0 or 16
#pragma unroll
    for (int r = 0; r < 8; ++r) {
      int krow = m * 16 + r + 8 * hi;
      float v = (krow < NKV) ? (acc[r] + bias) : 0.0f;
      if (pair == 0) kmat[(h * 64 + krow) * 32 + hdb + l16] = (__bf16)v;
      else           vmat[(h * 32 + hdb + l16) * 64 + krow] = (__bf16)v;
    }
  }
  __syncthreads();

  // ---- Phase 3: 7 groups of 32 query rows ----
  const float* qsrc = q + (size_t)b * QNQ * CDIM;
  float* outb = out + (size_t)b * QNQ * CDIM;

  for (int g = 0; g < GROUPS; ++g) {
    // 3a: stage q rows (pad rows -> 0)
    for (int i = tid; i < 32 * CDIM; i += 256) {
      int r = i / CDIM, c = i % CDIM;
      int qrow = g * 32 + r;
      qg[i] = (__bf16)((qrow < QNQ) ? qsrc[qrow * CDIM + c] : 0.0f);
    }
    __syncthreads();

    // 3b: qp tile = qg @ w_q + b_q  (32 x 384); 48 tasks, 6/wave
    for (int t = wave; t < 48; t += 8) {
      int mt = t / 24, n = t % 24;
      const __bf16* arow = qg + (mt * 16 + l16) * CDIM;
      const __bf16* brow = w_qT + (n * 16 + l16) * CDIM;
      v8f acc = {};
#pragma unroll
      for (int kc = 0; kc < 6; ++kc) {
        v16bf a  = *(const v16bf*)(arow + kc * 32 + hi * 16);
        v16bf bb = *(const v16bf*)(brow + kc * 32 + hi * 16);
        acc = wmma_bf16(a, bb, acc);
      }
      float bias = b_q[n * 16 + l16];
#pragma unroll
      for (int r = 0; r < 8; ++r)
        qpt[(mt * 16 + r + 8 * hi) * DIMC + n * 16 + l16] = (__bf16)(acc[r] + bias);
    }
    __syncthreads();

    // 3c: attention; 24 tasks (mtile x head), 3/wave
    for (int t = wave; t < 24; t += 8) {
      int mt = t / 12, h = t % 12;
      // S = qp_h (16x32) x K_h^T (32x64): 4 WMMAs
      v16bf a = *(const v16bf*)(qpt + (mt * 16 + l16) * DIMC + h * HDIM + hi * 16);
      v8f s[4];
#pragma unroll
      for (int nt = 0; nt < 4; ++nt) {
        v16bf bb = *(const v16bf*)(kmat + (h * 64 + nt * 16 + l16) * 32 + hi * 16);
        v8f z = {};
        s[nt] = wmma_bf16(a, bb, z);
      }
      // fragment-layout bias: 128 contiguous bytes per lane, coalesced
      const float* bp = biasPk + (((size_t)h * NTILE + (g * 2 + mt)) * 32 + lane) * 32;
      float bv[32];
#pragma unroll
      for (int j = 0; j < 32; ++j) bv[j] = bp[j];
      float p[4][8], rowmax[8], rowsum[8];
#pragma unroll
      for (int r = 0; r < 8; ++r) rowmax[r] = -3.0e38f;
#pragma unroll
      for (int nt = 0; nt < 4; ++nt)
#pragma unroll
        for (int r = 0; r < 8; ++r) {
          float v = s[nt][r] * SCALE_F + bv[nt * 8 + r];
          p[nt][r] = v;
          rowmax[r] = fmaxf(rowmax[r], v);
        }
#pragma unroll
      for (int r = 0; r < 8; ++r) {
        float m = rowmax[r];
        m = fmaxf(m, __shfl_xor(m, 1, 32));
        m = fmaxf(m, __shfl_xor(m, 2, 32));
        m = fmaxf(m, __shfl_xor(m, 4, 32));
        m = fmaxf(m, __shfl_xor(m, 8, 32));
        rowmax[r] = m;
        rowsum[r] = 0.0f;
      }
#pragma unroll
      for (int nt = 0; nt < 4; ++nt)
#pragma unroll
        for (int r = 0; r < 8; ++r) {
          float e = __expf(p[nt][r] - rowmax[r]);
          p[nt][r] = e;
          rowsum[r] += e;
        }
#pragma unroll
      for (int r = 0; r < 8; ++r) {
        float sm = rowsum[r];
        sm += __shfl_xor(sm, 1, 32);
        sm += __shfl_xor(sm, 2, 32);
        sm += __shfl_xor(sm, 4, 32);
        sm += __shfl_xor(sm, 8, 32);
        rowsum[r] = 1.0f / sm;
      }
      // P -> per-wave LDS buffer (re-fragment for P @ V)
      __bf16* pw = pbuf + wave * 1024;
#pragma unroll
      for (int nt = 0; nt < 4; ++nt)
#pragma unroll
        for (int r = 0; r < 8; ++r)
          pw[(r + 8 * hi) * 64 + nt * 16 + l16] = (__bf16)(p[nt][r] * rowsum[r]);
      // O = P (16x64) x V (64x32): 4 WMMAs
      v8f o0 = {}, o1 = {};
#pragma unroll
      for (int kc = 0; kc < 2; ++kc) {
        v16bf pa = *(const v16bf*)(pw + l16 * 64 + kc * 32 + hi * 16);
        v16bf v0 = *(const v16bf*)(vmat + (h * 32 + l16) * 64 + kc * 32 + hi * 16);
        v16bf v1 = *(const v16bf*)(vmat + (h * 32 + 16 + l16) * 64 + kc * 32 + hi * 16);
        o0 = wmma_bf16(pa, v0, o0);
        o1 = wmma_bf16(pa, v1, o1);
      }
#pragma unroll
      for (int r = 0; r < 8; ++r) {
        obuf[(mt * 16 + r + 8 * hi) * DIMC + h * HDIM + l16]      = (__bf16)o0[r];
        obuf[(mt * 16 + r + 8 * hi) * DIMC + h * HDIM + 16 + l16] = (__bf16)o1[r];
      }
    }
    __syncthreads();

    // 3d: out tile = obuf @ w_proj + b_proj  (32 x 192); 24 tasks, 3/wave
    for (int t = wave; t < 24; t += 8) {
      int mt = t / 12, n = t % 12;
      const __bf16* arow = obuf + (mt * 16 + l16) * DIMC;
      const __bf16* brow = w_projT + (n * 16 + l16) * DIMC;
      v8f acc = {};
#pragma unroll
      for (int kc = 0; kc < 12; ++kc) {
        v16bf a  = *(const v16bf*)(arow + kc * 32 + hi * 16);
        v16bf bb = *(const v16bf*)(brow + kc * 32 + hi * 16);
        acc = wmma_bf16(a, bb, acc);
      }
      float bias = b_proj[n * 16 + l16];
#pragma unroll
      for (int r = 0; r < 8; ++r) {
        int qrow = g * 32 + mt * 16 + r + 8 * hi;
        if (qrow < QNQ)
          outb[(size_t)qrow * CDIM + n * 16 + l16] = acc[r] + bias;
      }
    }
    __syncthreads();
  }
}

extern "C" void kernel_launch(void* const* d_in, const int* in_sizes, int n_in,
                              void* d_out, int out_size, void* d_ws, size_t ws_size,
                              hipStream_t stream) {
  (void)in_sizes; (void)n_in; (void)out_size; (void)ws_size;
  const float* kv         = (const float*)d_in[0];
  const float* q          = (const float*)d_in[1];
  const float* w_kv       = (const float*)d_in[2];
  const float* b_kv       = (const float*)d_in[3];
  const float* w_q        = (const float*)d_in[4];
  const float* b_q        = (const float*)d_in[5];
  const float* bias_table = (const float*)d_in[6];
  const float* w_proj     = (const float*)d_in[7];
  const float* b_proj     = (const float*)d_in[8];
  float* out = (float*)d_out;

  char* ws = (char*)d_ws;
  __bf16* w_kvT   = (__bf16*)(ws);            // 589824 B
  __bf16* w_qT    = (__bf16*)(ws + 589824);   // 147456 B
  __bf16* w_projT = (__bf16*)(ws + 737280);   // 147456 B
  float*  biasPk  = (float*)(ws + 884736);    // 688128 B  (total ~1.5 MB)

  const int preptot = KVOUT * DIMC + DIMC * CDIM + CDIM * DIMC + NHEAD * NTILE * 32 * 32;
  wca_prep_kernel<<<(preptot + 255) / 256, 256, 0, stream>>>(
      w_kv, w_q, w_proj, bias_table, w_kvT, w_qT, w_projT, biasPk);
  wca_main_kernel<<<BWIN, 256, 0, stream>>>(
      kv, q, b_kv, b_q, b_proj, w_kvT, w_qT, w_projT, biasPk, out);
}
